// Spike_LSTM_37632503447568
// MI455X (gfx1250) — matrix operand order repaired and checked by reading, not compiled, not run on hardware
//
#include <hip/hip_runtime.h>
#include <hip/hip_bf16.h>

// ---------------------------------------------------------------------------
// Spiking LSTM forward, 2 layers: L=1024, B=32, C=H=512, G=4H=2048.
//   init counters -> x to f16 -> 4x transpose weights to f16 [2048,512]
//   -> WMMA GEMM xp0 = x@Wih0+b0 (swizzled tile layout)
//   -> persistent WMMA scan layer0 (writes y0 f16)
//   -> WMMA GEMM xp1 = y0@Wih1+b1 -> persistent WMMA scan layer1 (y1 f32)
// GEMM: each wave computes a 16x64 strip; A fragment reused across 4 WMMAs;
// all 4 B fragments preloaded into distinct registers so load waits stagger.
// xp stored in WMMA-fragment order [mtile][ntile][lane][8f32] so both the
// GEMM epilogue and the scan's accumulator init are contiguous 128b accesses.
// Scan: 8 blocks x 8 waves; each wave owns a 16x16 (batch x hidden) tile and
// all 4 gate tiles; W_hh^T slices resident in 256KB LDS; h double-buffered in
// global f16; one atomic grid-sync per step.
// ---------------------------------------------------------------------------

typedef _Float16 half_t;
typedef __attribute__((ext_vector_type(16))) _Float16 v16h;
typedef __attribute__((ext_vector_type(8)))  _Float16 v8h;
typedef __attribute__((ext_vector_type(8)))  float    v8f;

#define Lc 1024
#define Bc 32
#define Cc 512
#define Hc 512
#define Gc 2048
#define Mrows (Lc * Bc)        // 32768
#define NTILES (Gc / 16)       // 128
#define SCAN_BLOCKS 8
#define SCAN_LDS (4 * 4 * 16 * 512 * 2)   // 256 KiB: [4 jh][4 gates][16 cols][512 k] f16

// ------------------------------ utility kernels -----------------------------

__global__ void k_init(unsigned* counters) {
    if (threadIdx.x < 2) counters[threadIdx.x] = 0u;
}

__global__ void k_f32_to_f16(const float* __restrict__ src, half_t* __restrict__ dst, int n) {
    int i = blockIdx.x * blockDim.x + threadIdx.x;
    if (i < n) dst[i] = (half_t)src[i];
}

// src [K,N] row-major f32 -> dst [N,K] row-major f16 (W^T, contiguous in K)
__global__ void k_transpose_to_f16(const float* __restrict__ src, half_t* __restrict__ dst,
                                   int K, int N) {
    int i = blockIdx.x * blockDim.x + threadIdx.x;
    if (i < K * N) {
        int k = i / N, n = i - k * N;
        dst[n * K + k] = (half_t)src[i];
    }
}

// ------------------------------- input GEMM ---------------------------------
// XPs (swizzled) = A[M,512](f16) * BT[2048,512](f16) + bias.
// grid = (32 n-groups of 4 tiles, M/16/8), block = 256 (8 waves).

__global__ __launch_bounds__(256) void k_gemm(const half_t* __restrict__ A,
                                              const half_t* __restrict__ BT,
                                              const float* __restrict__ bias,
                                              float* __restrict__ XPs) {
    const int lane = threadIdx.x & 31;
    const int wave = threadIdx.x >> 5;
    const int nt0  = blockIdx.x * 4;                 // first of 4 n-tiles
    const int mt   = blockIdx.y * 8 + wave;          // m-tile
    const int col  = lane & 15;
    const int hsel = lane >> 4;

    const half_t* Ap = A + (size_t)mt * 16 * Cc;
    const half_t* Bp = BT + (size_t)nt0 * 16 * Cc;

    v8f acc[4] = {};
#pragma unroll 2
    for (int kc = 0; kc < 16; ++kc) {
        const int kb = kc * 32;
        // A fragment: lane<16 -> K in {kb..kb+15}; lanes>=16 shifted by 16
        const half_t* ap = Ap + (lane & 15) * Cc + kb + hsel * 16;
        v8h alo = *(const v8h*)ap;
        v8h ahi = *(const v8h*)(ap + 8);
        v16h af;
#pragma unroll
        for (int i = 0; i < 8; ++i) { af[i] = alo[i]; af[8 + i] = ahi[i]; }
        // preload all 4 B fragments (distinct regs -> staggered load waits)
        v16h bf[4];
#pragma unroll
        for (int j = 0; j < 4; ++j)
            bf[j] = *(const v16h*)(Bp + (j * 16 + col) * Cc + kb + hsel * 16);
#pragma unroll
        for (int j = 0; j < 4; ++j)
            acc[j] = __builtin_amdgcn_wmma_f32_16x16x32_f16(false, af, false, bf[j],
                                                            (short)0, acc[j],
                                                            false, false);
    }
#pragma unroll
    for (int j = 0; j < 4; ++j) {
        const float bv = bias[(nt0 + j) * 16 + col];
#pragma unroll
        for (int r = 0; r < 8; ++r) acc[j][r] += bv;
        // fragment-order store: [mtile][ntile][lane][8] -> 32B contiguous per lane
        *(v8f*)(XPs + ((((size_t)mt * NTILES) + nt0 + j) * 32 + lane) * 8) = acc[j];
    }
}

// ------------------------------ recurrent scan ------------------------------

__global__ __launch_bounds__(256) void k_scan(const float* __restrict__ XPs,  // swizzled xp
                                              const half_t* __restrict__ UT,  // W_hh^T [2048,512] f16
                                              half_t* __restrict__ hbuf,      // [2, 32, 512] f16
                                              half_t* __restrict__ y16,       // [L*B, 512] or null
                                              float*  __restrict__ y32,       // [L*B, 512] or null
                                              float*  __restrict__ hout,      // [32, 512]
                                              float*  __restrict__ cout,      // [32, 512]
                                              unsigned* __restrict__ counter) {
    extern __shared__ half_t smem[];  // [4 jh][4 gate][16 cols][512 k]

    const int lane  = threadIdx.x & 31;
    const int wave  = threadIdx.x >> 5;
    const int jl    = wave >> 1;                  // local hidden-col tile 0..3
    const int jh    = blockIdx.x * 4 + jl;        // global hidden-col tile 0..31
    const int mt    = wave & 1;                   // batch tile 0..1
    const int col   = lane & 15;
    const int hsel  = lane >> 4;
    const int rbase = hsel * 8;
    const int row0  = mt * 16;

    // Stage this block's W_hh^T slices into LDS (once): 16 tiles of 16x512 f16.
    for (int i = threadIdx.x; i < 16384; i += 256) {        // 16B chunks
        const int tile = i >> 10;                           // jl*4 + gate
        const int cit  = i & 1023;
        const int c    = cit >> 6;
        const int kk   = (cit & 63) << 3;
        const int g    = tile & 3;
        const int jt   = tile >> 2;
        const half_t* s = UT + (size_t)(g * Hc + (blockIdx.x * 4 + jt) * 16 + c) * Cc + kk;
        *(v8h*)(smem + (tile * 16 + c) * Cc + kk) = *(const v8h*)s;
    }
    __syncthreads();

    const half_t* myU = smem + (jl * 4) * (16 * Cc);

    v8f cst   = {};   // c-state, WMMA D layout: (row0+rbase+r, jh*16+col)
    v8f hlast = {};

    for (int t = 0; t < Lc; ++t) {
        // fragment-order xp: mtile = t*2 + mt, ntile = g*32 + jh
        const float* xpt = XPs + ((((size_t)(t * 2 + mt)) * NTILES) * 32 + lane) * 8;
        if (t + 1 < Lc)
            __builtin_prefetch(XPs + ((((size_t)((t + 1) * 2 + mt)) * NTILES + jh) * 32) * 8,
                               0, 1);

        v8f acc[4];
#pragma unroll
        for (int g = 0; g < 4; ++g)
            acc[g] = *(const v8f*)(xpt + (size_t)(g * 32 + jh) * 32 * 8);

        if (t > 0) {  // h(t=0) == 0 -> skip recurrent matmul on first step
            const half_t* hprev = hbuf + ((t - 1) & 1) * (Bc * Hc) + row0 * Hc;
#pragma unroll 2
            for (int kc = 0; kc < 16; ++kc) {
                const int kb = kc * 32;
                const half_t* ap = hprev + (lane & 15) * Hc + kb + hsel * 16;
                v8h alo = *(const v8h*)ap;
                v8h ahi = *(const v8h*)(ap + 8);
                v16h af;
#pragma unroll
                for (int i = 0; i < 8; ++i) { af[i] = alo[i]; af[8 + i] = ahi[i]; }
                // preload all 4 gate B fragments from LDS (staggered ds waits)
                v16h bf[4];
#pragma unroll
                for (int g = 0; g < 4; ++g)
                    bf[g] = *(const v16h*)(myU + g * (16 * Cc) + col * Cc + kb + hsel * 16);
#pragma unroll
                for (int g = 0; g < 4; ++g)
                    acc[g] = __builtin_amdgcn_wmma_f32_16x16x32_f16(false, af, false, bf[g],
                                                                    (short)0, acc[g],
                                                                    false, false);
            }
        }

        // spike (heaviside) + cell update, all in registers
        v8f hnew;
#pragma unroll
        for (int r = 0; r < 8; ++r) {
            const float iv = acc[0][r] >= 0.f ? 1.f : 0.f;
            const float fv = acc[1][r] >= 0.f ? 1.f : 0.f;
            const float gv = acc[2][r] >= 0.f ? 1.f : 0.f;
            const float ov = acc[3][r] >= 0.f ? 1.f : 0.f;
            float cv = fv * cst[r] + iv * gv;
            cv = fminf(cv, 1.0f);
            cst[r] = cv;
            hnew[r] = ov * cv;
        }
        hlast = hnew;

        // publish h for next step (double-buffered)
        half_t* hw = hbuf + (t & 1) * (Bc * Hc) + row0 * Hc + jh * 16;
#pragma unroll
        for (int r = 0; r < 8; ++r) hw[(rbase + r) * Hc + col] = (half_t)hnew[r];

        if (y16) {
            half_t* yp = y16 + (size_t)(t * Bc + row0) * Hc + jh * 16;
#pragma unroll
            for (int r = 0; r < 8; ++r) yp[(rbase + r) * Hc + col] = (half_t)hnew[r];
        }
        if (y32) {
            float* yp = y32 + (size_t)(t * Bc + row0) * Hc + jh * 16;
#pragma unroll
            for (int r = 0; r < 8; ++r) yp[(rbase + r) * Hc + col] = hnew[r];
        }

        // grid-wide sync across the 8 persistent blocks
        __syncthreads();
        if (threadIdx.x == 0) {
            __threadfence();
            atomicAdd(counter, 1u);
            const unsigned tgt = (unsigned)(t + 1) * gridDim.x;
            while (__hip_atomic_load(counter, __ATOMIC_ACQUIRE,
                                     __HIP_MEMORY_SCOPE_AGENT) < tgt) {
                __builtin_amdgcn_s_sleep(1);
            }
            __threadfence();
        }
        __syncthreads();
    }

    // final states
#pragma unroll
    for (int r = 0; r < 8; ++r) {
        const int row = row0 + rbase + r;
        const int ci  = jh * 16 + col;
        hout[row * Hc + ci] = hlast[r];
        cout[row * Hc + ci] = cst[r];
    }
}

// --------------------------------- launcher ---------------------------------

extern "C" void kernel_launch(void* const* d_in, const int* in_sizes, int n_in,
                              void* d_out, int out_size, void* d_ws, size_t ws_size,
                              hipStream_t stream) {
    (void)in_sizes; (void)n_in; (void)out_size; (void)ws_size;

    const float* x    = (const float*)d_in[0];
    const float* Wih0 = (const float*)d_in[1];
    const float* Whh0 = (const float*)d_in[2];
    const float* b0   = (const float*)d_in[3];
    const float* Wih1 = (const float*)d_in[4];
    const float* Whh1 = (const float*)d_in[5];
    const float* b1   = (const float*)d_in[6];
    float* out = (float*)d_out;

    // workspace carve-out (256B aligned slices)
    char* ws = (char*)d_ws;
    size_t off = 0;
    auto carve = [&](size_t bytes) {
        char* p = ws + off;
        off = (off + bytes + 255) & ~(size_t)255;
        return p;
    };
    half_t*   xh   = (half_t*)carve((size_t)Mrows * Cc * 2);   // x as f16
    half_t*   W0T  = (half_t*)carve((size_t)Gc * Cc * 2);
    half_t*   U0T  = (half_t*)carve((size_t)Gc * Hc * 2);
    half_t*   W1T  = (half_t*)carve((size_t)Gc * Hc * 2);
    half_t*   U1T  = (half_t*)carve((size_t)Gc * Hc * 2);
    float*    xp   = (float*)carve((size_t)Mrows * Gc * 4);    // swizzled projections
    half_t*   y0   = (half_t*)carve((size_t)Mrows * Hc * 2);   // layer0 output (binary)
    half_t*   hbuf = (half_t*)carve((size_t)2 * Bc * Hc * 2);  // h double buffer
    unsigned* ctr  = (unsigned*)carve(256);

    // output layout: y1 [L,B,H], state_h [2,B,H], state_c [2,B,H]
    const size_t ST_H = (size_t)Lc * Bc * Hc;
    const size_t ST_C = ST_H + (size_t)2 * Bc * Hc;

    hipFuncSetAttribute((const void*)k_scan,
                        hipFuncAttributeMaxDynamicSharedMemorySize, SCAN_LDS);

    k_init<<<1, 64, 0, stream>>>(ctr);
    k_f32_to_f16<<<(Mrows * Cc + 255) / 256, 256, 0, stream>>>(x, xh, Mrows * Cc);
    k_transpose_to_f16<<<(Cc * Gc + 255) / 256, 256, 0, stream>>>(Wih0, W0T, Cc, Gc);
    k_transpose_to_f16<<<(Hc * Gc + 255) / 256, 256, 0, stream>>>(Whh0, U0T, Hc, Gc);
    k_transpose_to_f16<<<(Hc * Gc + 255) / 256, 256, 0, stream>>>(Wih1, W1T, Hc, Gc);
    k_transpose_to_f16<<<(Hc * Gc + 255) / 256, 256, 0, stream>>>(Whh1, U1T, Hc, Gc);

    // layer 0
    k_gemm<<<dim3(NTILES / 4, Mrows / 16 / 8), 256, 0, stream>>>(xh, W0T, b0, xp);
    k_scan<<<SCAN_BLOCKS, 256, SCAN_LDS, stream>>>(
        xp, U0T, hbuf, y0, nullptr,
        out + ST_H, out + ST_C, ctr + 0);

    // layer 1
    k_gemm<<<dim3(NTILES / 4, Mrows / 16 / 8), 256, 0, stream>>>(y0, W1T, b1, xp);
    k_scan<<<SCAN_BLOCKS, 256, SCAN_LDS, stream>>>(
        xp, U1T, hbuf, nullptr, out,
        out + ST_H + (size_t)Bc * Hc, out + ST_C + (size_t)Bc * Hc, ctr + 1);
}
